// NeuralAudioCodec_70231305224698
// MI455X (gfx1250) — matrix operand order repaired and checked by reading, not compiled, not run on hardware
//
#include <hip/hip_runtime.h>
#include <math.h>

// ---------------- types for WMMA ----------------
typedef __attribute__((ext_vector_type(16))) __bf16 v16bf;
typedef __attribute__((ext_vector_type(8)))  float  v8f;

union BF16x16 { unsigned short us[16]; v16bf v; };

__device__ __forceinline__ unsigned short f2bf(float f) {
  unsigned int u = __float_as_uint(f);
  u += 0x7FFFu + ((u >> 16) & 1u);       // round-to-nearest-even
  return (unsigned short)(u >> 16);
}

// ---- CDNA5 async-to-LDS helpers (ASYNCcnt-tracked DMA into LDS) ----
__device__ __forceinline__ void async_load_b128(unsigned lds_addr, const void* gaddr) {
  asm volatile("global_load_async_to_lds_b128 %0, %1, off"
               :: "v"(lds_addr), "v"((unsigned long long)gaddr)
               : "memory");
}
__device__ __forceinline__ void wait_async0() {
  asm volatile("s_wait_asynccnt 0x0" ::: "memory");
}
// LDS byte address = low 32 bits of a generic pointer to __shared__ (aperture rule)
__device__ __forceinline__ unsigned lds_addr_of(const void* p) {
  return (unsigned)(unsigned long long)p;
}

// ---------------- constants ----------------
#define BATCH 8
#define SLEN  65536
#define DHID  512
#define CBK   1024
#define NCB   8
#define TQ    4096                     // latent time steps
#define NROWS (BATCH * TQ)             // 32768 VQ rows
#define NE_F  16777216.0f              // 8*512*4096 elements for the mean

// =======================================================================
// weight / codebook packing to bf16
// =======================================================================
__global__ void pack_w_kernel(const float* __restrict__ w,
                              unsigned short* __restrict__ dst,
                              int Cout, int Cin, int transposeFlip) {
  int idx = blockIdx.x * 256 + threadIdx.x;
  int total = 15 * Cout * Cin;
  if (idx >= total) return;
  int ci = idx % Cin;
  int t  = idx / Cin;
  int co = t % Cout;
  int k  = t / Cout;
  float v;
  if (!transposeFlip) v = w[((size_t)co * Cin + ci) * 15 + k];          // enc [Cout][Cin][15]
  else                v = w[((size_t)ci * Cout + co) * 15 + (14 - k)];  // dec [Cin][Cout][15], flipped
  dst[idx] = f2bf(v);
}

__global__ void pack_cb_kernel(const float* __restrict__ cb,
                               unsigned short* __restrict__ dst, int total) {
  int idx = blockIdx.x * 256 + threadIdx.x;
  if (idx < total) dst[idx] = f2bf(cb[idx]);
}

__global__ void cnorm_kernel(const float* __restrict__ cb, float* __restrict__ cnorm) {
  int n = blockIdx.x;                              // over NCB*CBK entries
  const float* p = cb + (size_t)n * DHID;
  float s = 0.f;
  for (int d = threadIdx.x; d < DHID; d += 256) { float v = p[d]; s += v * v; }
  __shared__ float red[256];
  red[threadIdx.x] = s; __syncthreads();
  for (int o = 128; o > 0; o >>= 1) { if (threadIdx.x < o) red[threadIdx.x] += red[threadIdx.x + o]; __syncthreads(); }
  if (threadIdx.x == 0) cnorm[n] = red[0];
}

__global__ void zero_accum_kernel(float* p) {
  if (threadIdx.x == 0 && blockIdx.x == 0) *p = 0.f;
}

// =======================================================================
// encoder layer 0: 1 -> 64 channels, stride 2 (memory bound, scalar)
// =======================================================================
__global__ void enc0_kernel(const float* __restrict__ x, const float* __restrict__ w,
                            const float* __restrict__ bias, float* __restrict__ y) {
  int idx = blockIdx.x * 256 + threadIdx.x;        // [8][64][32768]
  int t  = idx & 32767;
  int co = (idx >> 15) & 63;
  int b  = idx >> 21;
  const float* xb = x + (size_t)b * SLEN;
  const float* wk = w + co * 15;
  float s = bias[co];
#pragma unroll
  for (int k = 0; k < 15; ++k) {
    int src = 2 * t + k - 7;
    if (src >= 0 && src < SLEN) s += wk[k] * xb[src];
  }
  y[idx] = s > 0.f ? s : __expf(s) - 1.f;
}

// =======================================================================
// generic implicit-GEMM WMMA conv (mode 0: stride-2 conv, mode 1: convT s2)
// Apk layout: [15][Cout][Cin] bf16.  One wave computes a 16(co)x16(t) tile.
// =======================================================================
__global__ void conv_wmma_kernel(const unsigned short* __restrict__ Apk,
                                 const float* __restrict__ bias,
                                 const float* __restrict__ x, float* __restrict__ y,
                                 int Cin, int Cout, int Lin, int Lout,
                                 int mode, int act) {
  const int lane = threadIdx.x & 31;
  const int wave = threadIdx.x >> 5;
  const int lm = lane & 15, hi = lane >> 4;
  const int b   = blockIdx.z;
  const int co0 = blockIdx.y * 16;
  const int t0  = blockIdx.x * 128 + wave * 16;
  const int t   = t0 + lm;                          // this lane's N column

  v8f acc = {};
  for (int k = 0; k < 15; ++k) {
    int s; bool valid;
    if (mode == 0) { s = 2 * t + k - 7; valid = (s >= 0) && (s < Lin); }
    else { int u = t + k - 7; s = u >> 1; valid = (u >= 0) && ((u & 1) == 0) && (s < Lin); }
    const unsigned short* arow = Apk + ((size_t)k * Cout + (co0 + lm)) * Cin;
    for (int ci0 = 0; ci0 < Cin; ci0 += 32) {
      BF16x16 a, bm;
#pragma unroll
      for (int h = 0; h < 8; ++h) a.us[h]     = arow[ci0 + hi * 16 + h];      // K = h + hi*16
#pragma unroll
      for (int h = 0; h < 8; ++h) a.us[8 + h] = arow[ci0 + hi * 16 + 8 + h];  // K = 8+h+hi*16
#pragma unroll
      for (int h = 0; h < 16; ++h) {
        int ci = ci0 + h + hi * 16;                                           // B: K = h + hi*16
        long ofs = ((long)b * Cin + ci) * (long)Lin + (long)s;
        bm.us[h] = valid ? f2bf(x[ofs]) : (unsigned short)0;
      }
      acc = __builtin_amdgcn_wmma_f32_16x16x32_bf16(false, a.v, false, bm.v,
                                                    (short)0, acc, false, false);
    }
  }
#pragma unroll
  for (int r = 0; r < 8; ++r) {
    int co = co0 + r + hi * 8;
    float v = acc[r] + bias[co];
    if (act == 1) v = v > 0.f ? v : __expf(v) - 1.f;
    y[((size_t)b * Cout + co) * Lout + t] = v;
  }
}

// =======================================================================
// z [8][512][4096] -> row-major residual rows [32768][512]; zero quantized
// =======================================================================
__global__ void t_fwd_kernel(const float* __restrict__ z,
                             float* __restrict__ rt, float* __restrict__ qt) {
  int idx = blockIdx.x * 256 + threadIdx.x;        // 2^24
  int t = idx & 4095;
  int d = (idx >> 12) & 511;
  int b = idx >> 21;
  float v = z[idx];
  size_t o = ((size_t)(b * TQ + t)) * DHID + d;
  rt[o] = v; qt[o] = 0.f;
}

__global__ void t_bwd_kernel(const float* __restrict__ qt, float* __restrict__ q) {
  int idx = blockIdx.x * 256 + threadIdx.x;
  int t = idx & 4095;
  int d = (idx >> 12) & 511;
  int b = idx >> 21;
  q[idx] = qt[((size_t)(b * TQ + t)) * DHID + d];
}

// =======================================================================
// VQ argmin with CDNA5 async-to-LDS double-buffered codebook staging.
// One wave = 16 residual rows; the whole block (8 waves / 128 rows) shares
// each 16-entry x 512-d codebook chunk (16KB) staged in LDS by the async
// DMA engine, overlapped with the 16-WMMA compute on the previous chunk.
// score = ||c||^2 - 2 z.c   (||z||^2 constant per row)
// =======================================================================
#define CHUNK_USHORTS (16 * DHID)          // 8192 ushorts = 16KB per chunk

__global__ void vq_argmin_kernel(const float* __restrict__ rt,
                                 const unsigned short* __restrict__ cbbf,
                                 const float* __restrict__ cnorm,
                                 int* __restrict__ codes) {
  const int tid  = threadIdx.x;
  const int lane = tid & 31, wave = tid >> 5;
  const int lm = lane & 15, hi = lane >> 4;
  const int row0 = blockIdx.x * 128 + wave * 16;
  const float* zrow = rt + (size_t)(row0 + lm) * DHID;

  __shared__ unsigned short cbuf[2][CHUNK_USHORTS];   // 32KB double buffer
  __shared__ float sv[8][32][8];                      // 8KB
  __shared__ int   sn[8][32][8];                      // 8KB

  // ---- hoist full 512-d A fragments (reused for all 1024 codebook cols) ----
  BF16x16 a[16];
#pragma unroll
  for (int j = 0; j < 16; ++j) {
    const float* p = zrow + j * 32;
#pragma unroll
    for (int h = 0; h < 8; ++h) a[j].us[h]     = f2bf(p[hi * 16 + h]);
#pragma unroll
    for (int h = 0; h < 8; ++h) a[j].us[8 + h] = f2bf(p[hi * 16 + 8 + h]);
  }

  float bestv[8]; int bestn[8];
#pragma unroll
  for (int r = 0; r < 8; ++r) { bestv[r] = 3.4e38f; bestn[r] = 0; }

  // ---- prologue: async-stage chunk 0 into buffer 0 (4 x b128 per thread) ----
#pragma unroll
  for (int r = 0; r < 4; ++r) {
    unsigned l = lds_addr_of(&cbuf[0][0]) + (unsigned)(tid * 16 + r * 4096);
    async_load_b128(l, (const char*)cbbf + tid * 16 + r * 4096);
  }

  for (int c = 0; c < 64; ++c) {
    wait_async0();          // my portion of chunk c has landed in LDS
    __syncthreads();        // everyone's portion landed; buf[(c+1)&1] readers done
    if (c + 1 < 64) {       // overlap: stage chunk c+1 while computing chunk c
      const char* g = (const char*)(cbbf + (size_t)(c + 1) * CHUNK_USHORTS);
      unsigned lb = lds_addr_of(&cbuf[(c + 1) & 1][0]);
#pragma unroll
      for (int r = 0; r < 4; ++r)
        async_load_b128(lb + (unsigned)(tid * 16 + r * 4096), g + tid * 16 + r * 4096);
    }

    const unsigned short* crow = &cbuf[c & 1][lm * DHID + hi * 16];
    v8f acc = {};
#pragma unroll
    for (int j = 0; j < 16; ++j) {
      BF16x16 bm;
#pragma unroll
      for (int h = 0; h < 16; ++h) bm.us[h] = crow[j * 32 + h];
      acc = __builtin_amdgcn_wmma_f32_16x16x32_bf16(false, a[j].v, false, bm.v,
                                                    (short)0, acc, false, false);
    }
    int n0 = c * 16;
    float cn = cnorm[n0 + lm];
#pragma unroll
    for (int r = 0; r < 8; ++r) {
      float score = cn - 2.f * acc[r];
      if (score < bestv[r]) { bestv[r] = score; bestn[r] = n0 + lm; }
    }
  }

  // ---- cross-lane argmin reduction (row M = r + hi*8, candidates over lm) ----
#pragma unroll
  for (int r = 0; r < 8; ++r) { sv[wave][lane][r] = bestv[r]; sn[wave][lane][r] = bestn[r]; }
  __syncthreads();
  if (lane < 16) {                                  // lane == local row
    int r = lane & 7, h2 = lane >> 3;
    float bv = 3.4e38f; int bn = 0x7fffffff;
    for (int j = 0; j < 16; ++j) {
      float v = sv[wave][h2 * 16 + j][r];
      int  n  = sn[wave][h2 * 16 + j][r];
      if (v < bv || (v == bv && n < bn)) { bv = v; bn = n; }
    }
    codes[row0 + lane] = bn;
  }
}

// =======================================================================
// gather z_q, accumulate quantized, update residual, accumulate loss
// =======================================================================
__global__ void vq_update_kernel(const float* __restrict__ cb,     // f32 [1024][512]
                                 const int* __restrict__ codes,
                                 float* __restrict__ rt, float* __restrict__ qt,
                                 float* __restrict__ loss_accum,
                                 float* __restrict__ codes_out) {
  int row = blockIdx.x;
  int code = codes[row];
  const float* c = cb + (size_t)code * DHID;
  size_t base = (size_t)row * DHID;
  float s = 0.f;
  for (int d = threadIdx.x; d < DHID; d += 256) {
    float zq = c[d];
    float r  = rt[base + d];
    float df = zq - r;
    s += df * df;
    qt[base + d] += zq;
    rt[base + d]  = r - zq;
  }
  __shared__ float red[256];
  red[threadIdx.x] = s; __syncthreads();
  for (int o = 128; o > 0; o >>= 1) { if (threadIdx.x < o) red[threadIdx.x] += red[threadIdx.x + o]; __syncthreads(); }
  if (threadIdx.x == 0) {
    atomicAdd(loss_accum, red[0]);
    codes_out[row] = (float)code;
  }
}

__global__ void finalize_loss_kernel(const float* __restrict__ accum, float* __restrict__ out) {
  if (threadIdx.x == 0 && blockIdx.x == 0) out[0] = accum[0] * (1.25f / NE_F);
}

// =======================================================================
// decoder layer 3: 64 -> 1 channels, convT s2, tanh (memory bound, scalar)
// =======================================================================
__global__ void dec3_kernel(const float* __restrict__ x,   // [8][64][32768]
                            const float* __restrict__ w,   // [64][1][15]
                            const float* __restrict__ bias,
                            float* __restrict__ y) {       // [8][65536]
  int idx = blockIdx.x * 256 + threadIdx.x;
  int t = idx & 65535;
  int b = idx >> 16;
  const float* xb = x + (size_t)b * 64 * 32768;
  float s = bias[0];
#pragma unroll
  for (int k = 0; k < 15; ++k) {
    int u = t + k - 7;
    if (u < 0 || (u & 1)) continue;
    int si = u >> 1;
    if (si >= 32768) continue;
    for (int ci = 0; ci < 64; ++ci)
      s += w[ci * 15 + (14 - k)] * xb[(size_t)ci * 32768 + si];
  }
  y[idx] = tanhf(s);
}

// =======================================================================
// host-side orchestration
// =======================================================================
extern "C" void kernel_launch(void* const* d_in, const int* in_sizes, int n_in,
                              void* d_out, int out_size, void* d_ws, size_t ws_size,
                              hipStream_t stream) {
  const float* audio = (const float*)d_in[0];
  const float* ew[4] = {(const float*)d_in[1], (const float*)d_in[3], (const float*)d_in[5], (const float*)d_in[7]};
  const float* eb[4] = {(const float*)d_in[2], (const float*)d_in[4], (const float*)d_in[6], (const float*)d_in[8]};
  const float* cbs   = (const float*)d_in[9];
  const float* dw[4] = {(const float*)d_in[10], (const float*)d_in[12], (const float*)d_in[14], (const float*)d_in[16]};
  const float* db[4] = {(const float*)d_in[11], (const float*)d_in[13], (const float*)d_in[15], (const float*)d_in[17]};
  float* out = (float*)d_out;

  // ---- workspace carve-up (all 256B aligned sizes) ----
  const size_t ACT = (size_t)16777216 * 4;         // 64MB activation tensors
  char* p = (char*)d_ws;
  float* bufA = (float*)p; p += ACT;
  float* bufB = (float*)p; p += ACT;
  float* rt   = (float*)p; p += ACT;               // residual rows [32768][512]
  float* qt   = (float*)p; p += ACT;               // quantized rows
  const int e_cout[3] = {128, 256, 512}, e_cin[3] = {64, 128, 256};
  const int d_cout[3] = {256, 128, 64},  d_cin[3] = {512, 256, 128};
  unsigned short* epk[3]; unsigned short* dpk[3];
  for (int l = 0; l < 3; ++l) { epk[l] = (unsigned short*)p; p += (size_t)15 * e_cout[l] * e_cin[l] * 2; }
  for (int l = 0; l < 3; ++l) { dpk[l] = (unsigned short*)p; p += (size_t)15 * d_cout[l] * d_cin[l] * 2; }
  unsigned short* cbbf = (unsigned short*)p; p += (size_t)NCB * CBK * DHID * 2;
  float* cnorm = (float*)p; p += (size_t)NCB * CBK * 4;
  int*   codes = (int*)p;  p += (size_t)NROWS * 4;
  float* accum = (float*)p; p += 256;

  // ---- pack weights / codebooks ----
  for (int l = 0; l < 3; ++l) {
    int tot = 15 * e_cout[l] * e_cin[l];
    pack_w_kernel<<<(tot + 255) / 256, 256, 0, stream>>>(ew[l + 1], epk[l], e_cout[l], e_cin[l], 0);
  }
  for (int l = 0; l < 3; ++l) {
    int tot = 15 * d_cout[l] * d_cin[l];
    pack_w_kernel<<<(tot + 255) / 256, 256, 0, stream>>>(dw[l], dpk[l], d_cout[l], d_cin[l], 1);
  }
  pack_cb_kernel<<<(NCB * CBK * DHID) / 256, 256, 0, stream>>>(cbs, cbbf, NCB * CBK * DHID);
  cnorm_kernel<<<NCB * CBK, 256, 0, stream>>>(cbs, cnorm);
  zero_accum_kernel<<<1, 64, 0, stream>>>(accum);

  // ---- encoder ----
  enc0_kernel<<<(BATCH * 64 * 32768) / 256, 256, 0, stream>>>(audio, ew[0], eb[0], bufA);
  conv_wmma_kernel<<<dim3(16384 / 128, 128 / 16, BATCH), 256, 0, stream>>>(epk[0], eb[1], bufA, bufB, 64, 128, 32768, 16384, 0, 1);
  conv_wmma_kernel<<<dim3(8192 / 128, 256 / 16, BATCH), 256, 0, stream>>>(epk[1], eb[2], bufB, bufA, 128, 256, 16384, 8192, 0, 1);
  conv_wmma_kernel<<<dim3(4096 / 128, 512 / 16, BATCH), 256, 0, stream>>>(epk[2], eb[3], bufA, bufB, 256, 512, 8192, 4096, 0, 1);

  // ---- residual VQ ----
  t_fwd_kernel<<<16777216 / 256, 256, 0, stream>>>(bufB, rt, qt);
  for (int i = 0; i < NCB; ++i) {
    vq_argmin_kernel<<<NROWS / 128, 256, 0, stream>>>(
        rt, cbbf + (size_t)i * CBK * DHID, cnorm + (size_t)i * CBK, codes);
    vq_update_kernel<<<NROWS, 256, 0, stream>>>(
        cbs + (size_t)i * CBK * DHID, codes, rt, qt, accum,
        out + 524289 + (size_t)i * NROWS);
  }
  finalize_loss_kernel<<<1, 64, 0, stream>>>(accum, out + 524288);

  // ---- decoder ----
  t_bwd_kernel<<<16777216 / 256, 256, 0, stream>>>(qt, bufA);
  conv_wmma_kernel<<<dim3(8192 / 128, 256 / 16, BATCH), 256, 0, stream>>>(dpk[0], db[0], bufA, bufB, 512, 256, 4096, 8192, 1, 1);
  conv_wmma_kernel<<<dim3(16384 / 128, 128 / 16, BATCH), 256, 0, stream>>>(dpk[1], db[1], bufB, bufA, 256, 128, 8192, 16384, 1, 1);
  conv_wmma_kernel<<<dim3(32768 / 128, 64 / 16, BATCH), 256, 0, stream>>>(dpk[2], db[2], bufA, bufB, 128, 64, 16384, 32768, 1, 1);
  dec3_kernel<<<(BATCH * SLEN) / 256, 256, 0, stream>>>(bufB, dw[3], db[3], out);

  (void)in_sizes; (void)n_in; (void)out_size; (void)ws_size;
}
